// LstmRnn_63015760167099
// MI455X (gfx1250) — matrix operand order repaired and checked by reading, not compiled, not run on hardware
//
#include <hip/hip_runtime.h>
#include <hip/hip_bf16.h>

// Problem constants (match reference)
#define BB 2048   // batch
#define TT 64     // warmup steps
#define FF 64     // input features
#define UU 512    // LSTM units
#define DD 128    // pred dim
#define LL 32     // latent
#define OO 24     // output steps
#define GG (4*UU) // gate width = 2048

typedef _Float16 half8 __attribute__((ext_vector_type(8)));
typedef _Float16 v16h  __attribute__((ext_vector_type(16)));
typedef float    v8f   __attribute__((ext_vector_type(8)));

// ---------------------------------------------------------------------------
// Fragment helpers for V_WMMA_F32_16X16X32_F16 (wave32 layouts per CDNA5 ISA)
// A (16x32 f16): lane L holds row M = L%16; v16h elems 0..7  = K (L/16)*8..+7
//                                            elems 8..15 = K 16+(L/16)*8..+7
// B (32x16 f16): lane L holds col N = L%16; same K mapping as A
// C (16x16 f32): lane L, vgpr v -> row v + 8*(L/16), col L%16
// ---------------------------------------------------------------------------
__device__ __forceinline__ v16h cat8(half8 lo, half8 hi) {
    return __builtin_shufflevector(lo, hi, 0,1,2,3,4,5,6,7,8,9,10,11,12,13,14,15);
}

// Accumulate acc[2][4] (32 rows x 64 cols wave tile) over K for one (A, W^T) pair.
// A: [M][lda] f16 row-major.  W: [N][K] f16 (pre-transposed weights).
__device__ __forceinline__ void mma_k_loop(const _Float16* __restrict__ A, int lda, int K,
                                           const _Float16* __restrict__ W,
                                           int m0, int n0, int lq, int lr,
                                           v8f (&acc)[2][4]) {
    for (int kb = 0; kb < K; kb += 32) {
        v16h a[2];
#pragma unroll
        for (int i = 0; i < 2; ++i) {
            const _Float16* ap = A + (size_t)(m0 + i*16 + lr) * lda + kb + lq*8;
            a[i] = cat8(*(const half8*)ap, *(const half8*)(ap + 16));
        }
#pragma unroll
        for (int j = 0; j < 4; ++j) {
            const _Float16* wp = W + (size_t)(n0 + j*16 + lr) * K + kb + lq*8;
            v16h b = cat8(*(const half8*)wp, *(const half8*)(wp + 16));
#pragma unroll
            for (int i = 0; i < 2; ++i)
                acc[i][j] = __builtin_amdgcn_wmma_f32_16x16x32_f16(
                    false, a[i], false, b, (short)0, acc[i][j], false, false);
        }
    }
}

// ---------------------------------------------------------------------------
// Plain dual-input GEMM:  C = A0 @ W0^T [+ A1 @ W1^T] + bias (optional relu).
// Each wave computes a 32x64 tile. Optional f32 store (ldc) / f16 store (ldch).
// ---------------------------------------------------------------------------
__global__ __launch_bounds__(256)
void gemm_wmma_f16(const _Float16* __restrict__ A0, int lda0, int K0,
                   const _Float16* __restrict__ W0,
                   const _Float16* __restrict__ A1, int lda1, int K1,
                   const _Float16* __restrict__ W1,
                   const float* __restrict__ bias,
                   float* __restrict__ C, int ldc,
                   _Float16* __restrict__ Ch, int ldch,
                   int M, int N, int relu) {
    const int gw   = (blockIdx.x * 256 + threadIdx.x) >> 5;
    const int lane = threadIdx.x & 31;
    const int ntN  = N >> 6;
    if (gw >= (M >> 5) * ntN) return;
    const int m0 = (gw / ntN) * 32;
    const int n0 = (gw % ntN) * 64;
    const int lq = lane >> 4;
    const int lr = lane & 15;

    v8f acc[2][4];
#pragma unroll
    for (int i = 0; i < 2; ++i)
#pragma unroll
        for (int j = 0; j < 4; ++j) { v8f z = {}; acc[i][j] = z; }

    mma_k_loop(A0, lda0, K0, W0, m0, n0, lq, lr, acc);
    if (K1 > 0)
        mma_k_loop(A1, lda1, K1, W1, m0, n0, lq, lr, acc);

#pragma unroll
    for (int i = 0; i < 2; ++i)
#pragma unroll
        for (int j = 0; j < 4; ++j) {
            const int col = n0 + j*16 + lr;
            const float bv = bias ? bias[col] : 0.0f;
#pragma unroll
            for (int v = 0; v < 8; ++v) {
                const int row = m0 + i*16 + lq*8 + v;
                float val = acc[i][j][v] + bv;
                if (relu) val = fmaxf(val, 0.0f);
                if (C)  C [(size_t)row * ldc  + col] = val;
                if (Ch) Ch[(size_t)row * ldch + col] = (_Float16)val;
            }
        }
}

// ---------------------------------------------------------------------------
// Fused LSTM step:  z = A0@W0^T + A1@W1^T + b  with gate-interleaved weights
// (GEMM column n = 4*u + gate, gate order i,f,g,o), then the LSTM cell update
// fused in the epilogue via per-wave LDS staging.  Writes c (in-place, f32)
// and h_out (f16).  h_in/h_out are distinct buffers (ping-pong).
// ---------------------------------------------------------------------------
__global__ __launch_bounds__(256)
void lstm_gemm_wmma(const _Float16* __restrict__ A0, int lda0, int K0,
                    const _Float16* __restrict__ W0,
                    const _Float16* __restrict__ A1, int lda1, int K1,
                    const _Float16* __restrict__ W1,
                    const float* __restrict__ bperm,   // [GG] gate-interleaved bias
                    float* __restrict__ c,             // [B][U] in/out
                    _Float16* __restrict__ hout) {     // [B][U] out
    __shared__ float stage[8 * 32 * 64];               // 8 waves x 32x64 f32 tile
    const int gw   = (blockIdx.x * 256 + threadIdx.x) >> 5;
    const int lane = threadIdx.x & 31;
    const int ntN  = GG >> 6;                          // 32 column panels
    if (gw >= (BB >> 5) * ntN) return;
    const int m0 = (gw / ntN) * 32;
    const int n0 = (gw % ntN) * 64;
    const int lq = lane >> 4;
    const int lr = lane & 15;

    v8f acc[2][4];
#pragma unroll
    for (int i = 0; i < 2; ++i)
#pragma unroll
        for (int j = 0; j < 4; ++j) { v8f z = {}; acc[i][j] = z; }

    mma_k_loop(A0, lda0, K0, W0, m0, n0, lq, lr, acc);
    mma_k_loop(A1, lda1, K1, W1, m0, n0, lq, lr, acc);

    // Stage tile (+bias) into this wave's private LDS slot.
    float* st = stage + (threadIdx.x >> 5) * (32 * 64);
#pragma unroll
    for (int i = 0; i < 2; ++i)
#pragma unroll
        for (int j = 0; j < 4; ++j) {
            const int cl = j*16 + lr;
            const float bv = bperm[n0 + cl];
#pragma unroll
            for (int v = 0; v < 8; ++v)
                st[(i*16 + lq*8 + v) * 64 + cl] = acc[i][j][v] + bv;
        }
    // Same-wave LDS write->read: DS ops are in-order per wave; wait DScnt only.
    asm volatile("s_wait_dscnt 0x0" ::: "memory");

    // Each lane owns one u (16 u's per tile, 2 lanes split the 32 rows).
    const int u = (n0 >> 2) + lr;                      // global unit index
#pragma unroll 4
    for (int it = 0; it < 16; ++it) {
        const int rl = lq * 16 + it;                   // local row (batch)
        const float* gz = st + rl * 64 + lr * 4;       // [i,f,g,o] contiguous
        const float zi = gz[0], zf = gz[1], zg = gz[2], zo = gz[3];
        const float ig = 1.0f / (1.0f + __expf(-zi));
        const float fg = 1.0f / (1.0f + __expf(-zf));
        const float og = 1.0f / (1.0f + __expf(-zo));
        const size_t idx = (size_t)(m0 + rl) * UU + u;
        const float cn = fg * c[idx] + ig * tanhf(zg);
        c[idx] = cn;
        hout[idx] = (_Float16)(og * tanhf(cn));
    }
}

// VAE bottleneck: z = zm + exp(0.5*zm)*eps,  zm = pred@Wl + bl    [B][L]
__global__ void vae_bottleneck(const float* __restrict__ pred, const float* __restrict__ Wl,
                               const float* __restrict__ bl, const float* __restrict__ eps,
                               float* __restrict__ zout, int n) {
    int idx = blockIdx.x * blockDim.x + threadIdx.x;
    if (idx >= n) return;
    int b = idx / LL, l = idx - b * LL;
    const float* p = pred + (size_t)b * DD;
    float acc = bl[l];
#pragma unroll 4
    for (int d = 0; d < DD; ++d) acc += p[d] * Wl[(size_t)d * LL + l];
    zout[idx] = acc + __expf(0.5f * acc) * eps[idx];
}

// Unbottleneck: pred = z@Wu + bu -> out[:,0,:] f32 and p16 f16 (decoder input)
__global__ void vae_unbottleneck(const float* __restrict__ zin, const float* __restrict__ Wu,
                                 const float* __restrict__ bu, float* __restrict__ out,
                                 _Float16* __restrict__ p16, int n) {
    int idx = blockIdx.x * blockDim.x + threadIdx.x;
    if (idx >= n) return;
    int b = idx / DD, d = idx - b * DD;
    const float* zb = zin + (size_t)b * LL;
    float acc = bu[d];
#pragma unroll
    for (int l = 0; l < LL; ++l) acc += zb[l] * Wu[(size_t)l * DD + d];
    out[(size_t)b * OO * DD + d] = acc;
    p16[(size_t)b * DD + d] = (_Float16)acc;
}

// W [K][N] f32 -> WT [N][K] f16
__global__ void transpose_f16(const float* __restrict__ W, _Float16* __restrict__ WT,
                              int K, int N) {
    int i = blockIdx.x * blockDim.x + threadIdx.x;
    if (i >= K * N) return;
    int k = i / N, n = i - k * N;
    WT[(size_t)n * K + k] = (_Float16)W[i];
}

// Gate weights W [K][4U] f32 -> WT [GG][K] f16, column n <- orig (n&3)*U + n>>2
__global__ void transpose_perm_f16(const float* __restrict__ W, _Float16* __restrict__ WT,
                                   int K) {
    int i = blockIdx.x * blockDim.x + threadIdx.x;
    if (i >= K * GG) return;
    int n = i / K, k = i - n * K;
    int corig = (n & 3) * UU + (n >> 2);
    WT[i] = (_Float16)W[(size_t)k * GG + corig];
}

__global__ void permute_bias(const float* __restrict__ b, float* __restrict__ bp) {
    int n = blockIdx.x * blockDim.x + threadIdx.x;
    if (n < GG) bp[n] = b[(n & 3) * UU + (n >> 2)];
}

__global__ void f32_to_f16(const float* __restrict__ s, _Float16* __restrict__ d, int n) {
    int i = blockIdx.x * blockDim.x + threadIdx.x;
    if (i < n) d[i] = (_Float16)s[i];
}
__global__ void zero_f16(_Float16* d, int n) {
    int i = blockIdx.x * blockDim.x + threadIdx.x;
    if (i < n) d[i] = (_Float16)0.0f;
}
__global__ void zero_f32(float* d, int n) {
    int i = blockIdx.x * blockDim.x + threadIdx.x;
    if (i < n) d[i] = 0.0f;
}

// ---------------------------------------------------------------------------
static inline void launch_gemm(hipStream_t s,
                               const _Float16* A0, int lda0, int K0, const _Float16* W0,
                               const _Float16* A1, int lda1, int K1, const _Float16* W1,
                               const float* bias, float* C, int ldc,
                               _Float16* Ch, int ldch, int M, int N, int relu) {
    int waves = (M >> 5) * (N >> 6);
    hipLaunchKernelGGL(gemm_wmma_f16, dim3((waves + 7) / 8), dim3(256), 0, s,
                       A0, lda0, K0, W0, A1, lda1, K1, W1,
                       bias, C, ldc, Ch, ldch, M, N, relu);
}

static inline void launch_lstm(hipStream_t s,
                               const _Float16* A0, int lda0, int K0, const _Float16* W0,
                               const _Float16* A1, int lda1, int K1, const _Float16* W1,
                               const float* bperm, float* c, _Float16* hout) {
    int waves = (BB >> 5) * (GG >> 6);   // 2048 waves -> 256 blocks
    hipLaunchKernelGGL(lstm_gemm_wmma, dim3(waves / 8), dim3(256), 0, s,
                       A0, lda0, K0, W0, A1, lda1, K1, W1, bperm, c, hout);
}

extern "C" void kernel_launch(void* const* d_in, const int* in_sizes, int n_in,
                              void* d_out, int out_size, void* d_ws, size_t ws_size,
                              hipStream_t stream) {
    const float* inputs = (const float*)d_in[0];
    const float* eps    = (const float*)d_in[1];
    const float* Wk_w   = (const float*)d_in[2];
    const float* Wr_w   = (const float*)d_in[3];
    const float* b_w    = (const float*)d_in[4];
    const float* Wk_d   = (const float*)d_in[5];
    const float* Wr_d   = (const float*)d_in[6];
    const float* b_d    = (const float*)d_in[7];
    const float* W1     = (const float*)d_in[8];
    const float* b1     = (const float*)d_in[9];
    const float* Wd     = (const float*)d_in[10];
    const float* bd     = (const float*)d_in[11];
    const float* Wl     = (const float*)d_in[12];
    const float* bl     = (const float*)d_in[13];
    const float* Wu     = (const float*)d_in[14];
    const float* bu     = (const float*)d_in[15];
    float* out = (float*)d_out;

    // --- workspace carve-out (256B aligned regions) ---
    char* p = (char*)d_ws;
    auto alloc = [&](size_t bytes) -> char* {
        char* r = p; p += (bytes + 255) & ~(size_t)255; return r;
    };
    _Float16* x16   = (_Float16*)alloc((size_t)BB*TT*FF*2);  // inputs as f16
    _Float16* WkwT  = (_Float16*)alloc((size_t)GG*FF*2);     // gate-permuted
    _Float16* WrwT  = (_Float16*)alloc((size_t)GG*UU*2);     // gate-permuted
    _Float16* WkdT  = (_Float16*)alloc((size_t)GG*DD*2);     // gate-permuted
    _Float16* WrdT  = (_Float16*)alloc((size_t)GG*UU*2);     // gate-permuted
    _Float16* W1T   = (_Float16*)alloc((size_t)UU*UU*2);
    _Float16* WdT   = (_Float16*)alloc((size_t)DD*UU*2);
    _Float16* hA    = (_Float16*)alloc((size_t)BB*UU*2);     // h ping
    _Float16* hB    = (_Float16*)alloc((size_t)BB*UU*2);     // h pong
    float*    cst   = (float*)   alloc((size_t)BB*UU*4);
    float*    bwp   = (float*)   alloc((size_t)GG*4);        // permuted b_w
    float*    bdp   = (float*)   alloc((size_t)GG*4);        // permuted b_d
    _Float16* x1    = (_Float16*)alloc((size_t)BB*UU*2);
    float*    pred  = (float*)   alloc((size_t)BB*DD*4);
    _Float16* p16   = (_Float16*)alloc((size_t)BB*DD*2);
    float*    zbuf  = (float*)   alloc((size_t)BB*LL*4);

    const int TB = 256;
    auto blocks = [](int n) { return (n + 255) / 256; };

    // --- precision conversion / weight transposes (tiny vs GEMM work) ---
    hipLaunchKernelGGL(f32_to_f16, dim3(blocks(BB*TT*FF)), dim3(TB), 0, stream, inputs, x16, BB*TT*FF);
    hipLaunchKernelGGL(transpose_perm_f16, dim3(blocks(FF*GG)), dim3(TB), 0, stream, Wk_w, WkwT, FF);
    hipLaunchKernelGGL(transpose_perm_f16, dim3(blocks(UU*GG)), dim3(TB), 0, stream, Wr_w, WrwT, UU);
    hipLaunchKernelGGL(transpose_perm_f16, dim3(blocks(DD*GG)), dim3(TB), 0, stream, Wk_d, WkdT, DD);
    hipLaunchKernelGGL(transpose_perm_f16, dim3(blocks(UU*GG)), dim3(TB), 0, stream, Wr_d, WrdT, UU);
    hipLaunchKernelGGL(transpose_f16, dim3(blocks(UU*UU)), dim3(TB), 0, stream, W1, W1T, UU, UU);
    hipLaunchKernelGGL(transpose_f16, dim3(blocks(UU*DD)), dim3(TB), 0, stream, Wd, WdT, UU, DD);
    hipLaunchKernelGGL(permute_bias, dim3(blocks(GG)), dim3(TB), 0, stream, b_w, bwp);
    hipLaunchKernelGGL(permute_bias, dim3(blocks(GG)), dim3(TB), 0, stream, b_d, bdp);
    hipLaunchKernelGGL(zero_f16, dim3(blocks(BB*UU)), dim3(TB), 0, stream, hA, BB*UU);
    hipLaunchKernelGGL(zero_f32, dim3(blocks(BB*UU)), dim3(TB), 0, stream, cst, BB*UU);

    // --- warmup encoder LSTM: 64 fused GEMM+cell steps (h ping-pong) ---
    _Float16* hcur = hA;
    _Float16* hnxt = hB;
    for (int t = 0; t < TT; ++t) {
        launch_lstm(stream,
                    x16 + (size_t)t*FF, TT*FF, FF, WkwT,   // x_t @ Wk_w (perm)
                    hcur, UU, UU, WrwT,                    // + h @ Wr_w (perm)
                    bwp, cst, hnxt);
        _Float16* tmp = hcur; hcur = hnxt; hnxt = tmp;
    }

    // --- dense head: x1 = relu(h@W1+b1);  pred = x1@Wd+bd ---
    launch_gemm(stream, hcur, UU, UU, W1T, nullptr, 0, 0, nullptr,
                b1, nullptr, 0, x1, UU, BB, UU, 1);
    launch_gemm(stream, x1, UU, UU, WdT, nullptr, 0, 0, nullptr,
                bd, pred, DD, nullptr, 0, BB, DD, 0);

    // --- VAE bottleneck + unbottleneck (writes out[:,0,:] and decoder input) ---
    hipLaunchKernelGGL(vae_bottleneck, dim3(blocks(BB*LL)), dim3(TB), 0, stream,
                       pred, Wl, bl, eps, zbuf, BB*LL);
    hipLaunchKernelGGL(vae_unbottleneck, dim3(blocks(BB*DD)), dim3(TB), 0, stream,
                       zbuf, Wu, bu, out, p16, BB*DD);

    // --- autoregressive decode: steps 1..23 ---
    for (int s = 1; s < OO; ++s) {
        launch_lstm(stream,
                    p16, DD, DD, WkdT,                     // p @ Wk_d (perm)
                    hcur, UU, UU, WrdT,                    // + h @ Wr_d (perm)
                    bdp, cst, hnxt);
        _Float16* tmp = hcur; hcur = hnxt; hnxt = tmp;
        launch_gemm(stream, hcur, UU, UU, W1T, nullptr, 0, 0, nullptr,
                    b1, nullptr, 0, x1, UU, BB, UU, 1);
        // pred_s -> out[:, s, :] (f32, row stride OO*DD) and p16 (next input)
        launch_gemm(stream, x1, UU, UU, WdT, nullptr, 0, 0, nullptr,
                    bd, out + (size_t)s*DD, OO*DD, p16, DD, BB, DD, 0);
    }
    (void)in_sizes; (void)n_in; (void)out_size; (void)ws_size;
}
